// AttentionCross_59691455480664
// MI455X (gfx1250) — matrix operand order repaired and checked by reading, not compiled, not run on hardware
//
#include <hip/hip_runtime.h>
#include <hip/hip_bf16.h>

#define HEADS 4
#define DIM_HEAD 32
#define Q_SCALE 0.17677669529663687f   // 32^-0.5

typedef __attribute__((ext_vector_type(16))) __bf16 v16bf;
typedef __attribute__((ext_vector_type(8)))  __bf16 v8bf;
typedef __attribute__((ext_vector_type(8)))  float  v8f;

union V16 { v16bf v; struct { v8bf lo; v8bf hi; } h; };

__device__ __forceinline__ unsigned short f32_to_bf16(float f) {
  union { float f; unsigned u; } x; x.f = f;
  unsigned u = x.u;
  unsigned r = u + 0x7FFFu + ((u >> 16) & 1u);   // round-to-nearest-even
  return (unsigned short)(r >> 16);
}

__device__ __forceinline__ void wait_async0() {
#if __has_builtin(__builtin_amdgcn_s_wait_asynccnt)
  __builtin_amdgcn_s_wait_asynccnt(0);
#else
  asm volatile("s_wait_asynccnt 0" ::: "memory");
#endif
}

// ---------------------------------------------------------------------------
// Mixed-precision GEMM:  C[b] = A[b](bf16, MxK row-major) * B[b] (+bias)
//   BT==0: B f32, element (k,n) at Bf[k*ldb + n]     (K-major rows: x, q)
//   BT==1: B f32, element (k,n) at Bf[n*ldb + k]     (N-major rows: context)
//   BT==2: B bf16, element (k,n) at Bh[n*ldb + k]    (N-major rows: qT) —
//          staged with global_load_async_to_lds_b128 (ASYNCcnt).
// Block: 256 threads = 8 waves; block tile M x 64; wave tile 16 x 64.
// Rows >= row16_0 are emitted as bf16 into C16 (used to produce v in bf16).
// ---------------------------------------------------------------------------
template<int BT>
__global__ __launch_bounds__(256) void gemm_bf16_kernel(
    const unsigned short* __restrict__ A, long strideA,
    const float* __restrict__ Bf, const unsigned short* __restrict__ Bh,
    int ldb, long strideB,
    float* __restrict__ C, long strideC,
    unsigned short* __restrict__ C16, long strideC16, int row16_0,
    const float* __restrict__ bias,
    int M, int N, int K)
{
  __shared__ __align__(16) unsigned short ldsB[64][40];  // [n][k], padded rows

  const int tid  = threadIdx.x;
  const int wave = tid >> 5;
  const int lane = tid & 31;
  const int bb   = blockIdx.z;
  const int n0   = blockIdx.x * 64;

  const unsigned short* Ab  = A + (long)bb * strideA;
  const float*          Bfb = (BT != 2) ? (Bf + (long)bb * strideB) : nullptr;
  const unsigned short* Bhb = (BT == 2) ? (Bh + (long)bb * strideB) : nullptr;
  float*                Cb  = C + (long)bb * strideC;
  unsigned short*       C16b = C16 ? (C16 + (long)bb * strideC16) : nullptr;

  const int mtiles = M >> 7;          // m-tiles per wave: M/128 (1 or 2)

  v8f acc[2][4];
#pragma unroll
  for (int i = 0; i < 2; ++i)
#pragma unroll
    for (int j = 0; j < 4; ++j) acc[i][j] = {};

  const int lrow   = lane & 15;
  const int khalfA = (lane < 16) ? 0 : 8;    // 16-bit A layout K split
  const int khalfB = (lane < 16) ? 0 : 8;    // 16-bit B layout K split

  for (int k0 = 0; k0 < K; k0 += 32) {
    // ---- stage 32(K) x 64(N) B tile into LDS as bf16, [n][k] ----
    if (BT == 0) {
      const int n  = tid & 63;
      const int kk = tid >> 6;                       // 0..3
#pragma unroll
      for (int r = 0; r < 8; ++r) {
        const int k = kk + r * 4;
        const float* p = Bfb + (long)(k0 + k) * ldb + n0 + n;
        ldsB[n][k] = f32_to_bf16(*p);
      }
      if (k0 + 32 < K)
        __builtin_prefetch(Bfb + (long)(k0 + 32 + kk) * ldb + n0 + n, 0, 1);
    } else if (BT == 1) {
      const int n  = tid >> 2;                       // 0..63
      const int kk = (tid & 3) * 8;
      const float* p = Bfb + (long)(n0 + n) * ldb + k0 + kk;
#pragma unroll
      for (int r = 0; r < 8; ++r) ldsB[n][kk + r] = f32_to_bf16(p[r]);
      if (k0 + 32 < K) __builtin_prefetch(p + 32, 0, 1);
    } else {
      // bf16 source: pure byte move -> async copy global -> LDS (ASYNCcnt)
      const int n  = tid >> 2;                       // 0..63
      const int kk = (tid & 3) * 8;                  // 8 bf16 = 16 bytes
      const unsigned short* p = Bhb + (long)(n0 + n) * ldb + k0 + kk;
      const unsigned lds_addr = (unsigned)(size_t)&ldsB[n][kk];
      asm volatile("global_load_async_to_lds_b128 %0, %1, off"
                   :: "v"(lds_addr), "v"((unsigned long long)(size_t)p)
                   : "memory");
      wait_async0();
    }
    __syncthreads();

    // ---- B fragments: lane holds N = nt*16+lrow, K = khalfB + 0..7, khalfB + 16..23 ----
    V16 bfr[4];
#pragma unroll
    for (int nt = 0; nt < 4; ++nt) {
      bfr[nt].h.lo = *(const v8bf*)&ldsB[nt * 16 + lrow][khalfB];
      bfr[nt].h.hi = *(const v8bf*)&ldsB[nt * 16 + lrow][khalfB + 16];
    }

#pragma unroll
    for (int i = 0; i < 2; ++i) {
      if (i < mtiles) {
        const int m = (wave + i * 8) * 16 + lrow;
        const unsigned short* ap = Ab + (long)m * K + k0 + khalfA;
        V16 af;
        af.h.lo = *(const v8bf*)ap;          // K = khalfA + 0..7
        af.h.hi = *(const v8bf*)(ap + 16);   // K = khalfA + 16..23
#pragma unroll
        for (int nt = 0; nt < 4; ++nt)
          acc[i][nt] = __builtin_amdgcn_wmma_f32_16x16x32_bf16(
              false, af.v, false, bfr[nt].v, (short)0, acc[i][nt], false, false);
      }
    }
    __syncthreads();
  }

  // ---- write out (C/D layout: VGPR r -> M=r / r+8; lane -> N) ----
#pragma unroll
  for (int i = 0; i < 2; ++i) {
    if (i < mtiles) {
      const int mbase = (wave + i * 8) * 16 + ((lane < 16) ? 0 : 8);
#pragma unroll
      for (int r = 0; r < 8; ++r) {
        const int row = mbase + r;
        const float vb = bias ? bias[row] : 0.0f;
#pragma unroll
        for (int nt = 0; nt < 4; ++nt) {
          const int col = n0 + nt * 16 + lrow;
          const float val = acc[i][nt][r] + vb;
          if (row < row16_0) Cb[(long)row * N + col] = val;
          else               C16b[(long)(row - row16_0) * N + col] = f32_to_bf16(val);
        }
      }
    }
  }
}

// ---------------------------------------------------------------------------
// softmax over d (32) for q, fused *SCALE; emits transposed bf16 qT[b][n][hd]
// so the final GEMM's B operand is bf16 with K contiguous.
// ---------------------------------------------------------------------------
__global__ __launch_bounds__(256) void softmax_q_kernel(
    const float* __restrict__ q, unsigned short* __restrict__ qT, int N)
{
  const int n = blockIdx.x * 256 + threadIdx.x;
  const int h = blockIdx.y, b = blockIdx.z;
  const float* p = q + ((long)(b * HEADS + h) * DIM_HEAD) * N + n;

  float v[DIM_HEAD];
  float mx = -1e30f;
#pragma unroll
  for (int d = 0; d < DIM_HEAD; ++d) { v[d] = p[(long)d * N]; mx = fmaxf(mx, v[d]); }
  float s = 0.0f;
#pragma unroll
  for (int d = 0; d < DIM_HEAD; ++d) { v[d] = __expf(v[d] - mx); s += v[d]; }
  const float inv = Q_SCALE / s;

  unsigned short* o = qT + ((long)b * N + n) * 128 + h * DIM_HEAD;
#pragma unroll
  for (int d = 0; d < DIM_HEAD; ++d) o[d] = f32_to_bf16(v[d] * inv);
}

// ---------------------------------------------------------------------------
// softmax over n (16384) for k rows: reads f32 k, writes bf16 k (same layout).
// ---------------------------------------------------------------------------
__global__ __launch_bounds__(256) void softmax_k_kernel(
    const float* __restrict__ kf, unsigned short* __restrict__ kb, int N)
{
  __shared__ float red[256];
  const int r = blockIdx.x, b = blockIdx.y;
  const float* p = kf + ((long)b * 128 + r) * N;
  unsigned short* o = kb + ((long)b * 128 + r) * N;
  const int tid = threadIdx.x;

  float mx = -1e30f;
  for (int n = tid; n < N; n += 256) mx = fmaxf(mx, p[n]);
  red[tid] = mx; __syncthreads();
  for (int s = 128; s > 0; s >>= 1) {
    if (tid < s) red[tid] = fmaxf(red[tid], red[tid + s]);
    __syncthreads();
  }
  mx = red[0]; __syncthreads();

  float sum = 0.0f;
  for (int n = tid; n < N; n += 256) sum += __expf(p[n] - mx);
  red[tid] = sum; __syncthreads();
  for (int s = 128; s > 0; s >>= 1) {
    if (tid < s) red[tid] += red[tid + s];
    __syncthreads();
  }
  const float inv = 1.0f / red[0];

  for (int n = tid; n < N; n += 256) o[n] = f32_to_bf16(__expf(p[n] - mx) * inv);
}

// ---------------------------------------------------------------------------
// ctx[b,h](32x32) = k(32xN) * v^T(Nx32); k,v already bf16, K contiguous ->
// fragments load straight from global, no conversion. 8 waves split K.
// ---------------------------------------------------------------------------
__global__ __launch_bounds__(256) void ctx_kernel(
    const unsigned short* __restrict__ kb, const unsigned short* __restrict__ vb,
    float* __restrict__ ctxout, int N)
{
  __shared__ float cs[32][33];
  const int tid = threadIdx.x;
  const int wave = tid >> 5;
  const int lane = tid & 31;
  const int h = blockIdx.x, b = blockIdx.y;

  const unsigned short* kp = kb + ((long)b * 128 + h * DIM_HEAD) * N;
  const unsigned short* vp = vb + ((long)b * 128 + h * DIM_HEAD) * N;

  for (int i = tid; i < 32 * 33; i += 256) (&cs[0][0])[i] = 0.0f;
  __syncthreads();

  const int lrow   = lane & 15;
  const int khalfA = (lane < 16) ? 0 : 8;
  const int khalfB = (lane < 16) ? 0 : 8;

  v8f acc[2][2];
  acc[0][0] = {}; acc[0][1] = {}; acc[1][0] = {}; acc[1][1] = {};

  const int kslice = N >> 3;            // 2048 per wave
  const long kbeg  = (long)wave * kslice;

  for (int kk = 0; kk < kslice; kk += 32) {
    const long k0 = kbeg + kk;
    V16 af[2], bfr[2];
#pragma unroll
    for (int t = 0; t < 2; ++t) {
      const unsigned short* ap = kp + (long)(t * 16 + lrow) * N + k0 + khalfA;
      af[t].h.lo = *(const v8bf*)ap;
      af[t].h.hi = *(const v8bf*)(ap + 16);
      const unsigned short* bp = vp + (long)(t * 16 + lrow) * N + k0 + khalfB;
      bfr[t].h.lo = *(const v8bf*)bp;
      bfr[t].h.hi = *(const v8bf*)(bp + 16);
    }
#pragma unroll
    for (int mi = 0; mi < 2; ++mi)
#pragma unroll
      for (int ni = 0; ni < 2; ++ni)
        acc[mi][ni] = __builtin_amdgcn_wmma_f32_16x16x32_bf16(
            false, af[mi].v, false, bfr[ni].v, (short)0, acc[mi][ni], false, false);
  }

#pragma unroll
  for (int mi = 0; mi < 2; ++mi)
#pragma unroll
    for (int ni = 0; ni < 2; ++ni)
#pragma unroll
      for (int r = 0; r < 8; ++r) {
        const int d = mi * 16 + ((lane < 16) ? r : r + 8);
        const int e = ni * 16 + lrow;
        atomicAdd(&cs[d][e], acc[mi][ni][r]);
      }
  __syncthreads();

  float* outp = ctxout + (long)(b * HEADS + h) * 32 * 32;
  for (int i = tid; i < 1024; i += 256) outp[i] = cs[i >> 5][i & 31];
}

// ---------------------------------------------------------------------------
// M2[b][o][h*32+d] = sum_e Wo[o][h*32+e] * ctx[b,h,d,e]  -> bf16
// ---------------------------------------------------------------------------
__global__ __launch_bounds__(128) void m2_kernel(
    const float* __restrict__ Wo, const float* __restrict__ ctx,
    unsigned short* __restrict__ M2)
{
  const int hd = threadIdx.x;           // 0..127
  const int o  = blockIdx.x;            // 0..255
  const int b  = blockIdx.y;            // 0..7
  const int h = hd >> 5, d = hd & 31;
  const float* wo = Wo + (long)o * 128 + h * 32;
  const float* cx = ctx + ((long)(b * HEADS + h) * 32 + d) * 32;
  float s = 0.0f;
#pragma unroll
  for (int e = 0; e < 32; ++e) s += wo[e] * cx[e];
  M2[((long)b * 256 + o) * 128 + hd] = f32_to_bf16(s);
}

// ---------------------------------------------------------------------------
__global__ __launch_bounds__(256) void cvt_f32_bf16_kernel(
    const float* __restrict__ src, unsigned short* __restrict__ dst, int n)
{
  const int i = blockIdx.x * 256 + threadIdx.x;
  if (i < n) dst[i] = f32_to_bf16(src[i]);
}

// ---------------------------------------------------------------------------
extern "C" void kernel_launch(void* const* d_in, const int* in_sizes, int n_in,
                              void* d_out, int out_size, void* d_ws, size_t ws_size,
                              hipStream_t stream) {
  const float* x    = (const float*)d_in[0];   // (8,256,16384)
  const float* ctxi = (const float*)d_in[1];   // (8,16384,512)
  const float* Wq   = (const float*)d_in[2];   // (128,256)
  const float* Wkv  = (const float*)d_in[3];   // (256,512)
  const float* Wo   = (const float*)d_in[4];   // (256,128)
  const float* bo   = (const float*)d_in[5];   // (256,)
  float* out = (float*)d_out;                  // (8,256,16384)

  const int B = 8, DIM = 256, N = 16384, HID = 128, CD = 512;

  char* ws = (char*)d_ws;
  float* q   = (float*)ws;                       ws += (size_t)B * HID * N * 4;  // 64 MB
  float* kf  = (float*)ws;                       ws += (size_t)B * HID * N * 4;  // 64 MB
  unsigned short* qT  = (unsigned short*)ws;     ws += (size_t)B * N * HID * 2;  // 32 MB
  unsigned short* kb  = (unsigned short*)ws;     ws += (size_t)B * HID * N * 2;  // 32 MB
  unsigned short* vb  = (unsigned short*)ws;     ws += (size_t)B * HID * N * 2;  // 32 MB
  float* cx = (float*)ws;                        ws += (size_t)B * HEADS * 32 * 32 * 4;
  unsigned short* m2     = (unsigned short*)ws;  ws += (size_t)B * 256 * 128 * 2;
  unsigned short* wq_bf  = (unsigned short*)ws;  ws += (size_t)HID * DIM * 2;
  unsigned short* wkv_bf = (unsigned short*)ws;

  // weights -> bf16
  cvt_f32_bf16_kernel<<<(HID * DIM + 255) / 256, 256, 0, stream>>>(Wq, wq_bf, HID * DIM);
  cvt_f32_bf16_kernel<<<(256 * CD + 255) / 256, 256, 0, stream>>>(Wkv, wkv_bf, 256 * CD);

  // q = Wq * x          (M=128, K=256; B f32, K-major rows)
  gemm_bf16_kernel<0><<<dim3(N / 64, 1, B), 256, 0, stream>>>(
      wq_bf, 0L, x, nullptr, N, (long)DIM * N,
      q, (long)HID * N, nullptr, 0L, 128, nullptr, HID, N, DIM);

  // kv = Wkv * ctx^T    (M=256, K=512; B f32, N-major rows)
  // rows 0..127 -> kf (f32, pre-softmax), rows 128..255 -> vb (bf16 directly)
  gemm_bf16_kernel<1><<<dim3(N / 64, 1, B), 256, 0, stream>>>(
      wkv_bf, 0L, ctxi, nullptr, CD, (long)N * CD,
      kf, (long)HID * N, vb, (long)HID * N, 128, nullptr, 256, N, CD);

  // softmaxes (emit bf16 operands for the WMMA consumers)
  softmax_q_kernel<<<dim3(N / 256, HEADS, B), 256, 0, stream>>>(q, qT, N);
  softmax_k_kernel<<<dim3(HID, B), 256, 0, stream>>>(kf, kb, N);

  // ctx = k * v^T per (b,h)   (pure bf16 WMMA, no conversions)
  ctx_kernel<<<dim3(HEADS, B), 256, 0, stream>>>(kb, vb, cx, N);

  // fold Wo into ctx
  m2_kernel<<<dim3(256, B), 128, 0, stream>>>(Wo, cx, m2);

  // out = M2[b] * qT^T + bo   (M=256, K=128; B bf16, async-to-LDS staging)
  gemm_bf16_kernel<2><<<dim3(N / 64, 1, B), 256, 0, stream>>>(
      m2, (long)256 * 128, nullptr, qT, HID, (long)N * HID,
      out, (long)DIM * N, nullptr, 0L, 256, bo, DIM, N, HID);
}
